// PraxisMixtureOfDepths_55903294325096
// MI455X (gfx1250) — matrix operand order repaired and checked by reading, not compiled, not run on hardware
//
#include <hip/hip_runtime.h>
#include <math.h>

// ---------------- problem constants (fixed by the reference) ----------------
#define BATCH 4
#define SEQ   4096
#define DMODEL 1024
#define DFF   4096
#define KTOP  2048            // SEQ * 0.5
#define MTOT  (BATCH * KTOP)  // 8192 selected rows total

// ---------------- WMMA types ----------------
typedef __attribute__((ext_vector_type(16))) __bf16 v16bf;
typedef __attribute__((ext_vector_type(8)))  float  v8f;

union Frag16 {
  v16bf v;
  uint4 q[2];
};

// float -> bf16 with round-to-nearest-even
__device__ __forceinline__ unsigned short f2bf(float f) {
  unsigned u = __float_as_uint(f);
  unsigned r = u + 0x7FFFu + ((u >> 16) & 1u);
  return (unsigned short)(r >> 16);
}

// ---- A fragment: row-major [M x K] bf16, 16x32 tile at (mBase, kBase) ----
// lane<16 holds row (lane), K = {kBase+0..7, kBase+16..23}
// lane>=16 holds row (lane-16), K = {kBase+8..15, kBase+24..31}
__device__ __forceinline__ v16bf load_a_frag(const unsigned short* __restrict__ base,
                                             int ld, int mBase, int kBase, int lane) {
  int r  = lane & 15;
  int hi = lane >> 4;
  const unsigned short* p = base + (size_t)(mBase + r) * ld + kBase + hi * 8;
  Frag16 f;
  f.q[0] = *(const uint4*)(p);        // K offsets hi*8 .. hi*8+7
  f.q[1] = *(const uint4*)(p + 16);   // K offsets 16+hi*8 .. 16+hi*8+7
  return f.v;
}

// ---- B fragment: B is [K x N]; we store it transposed as BT [N x K] row-major.
// lane<16: column n=lane, K = {kBase+0..7, kBase+16..23} ; lane>=16: column n=lane-16, K = {kBase+8..15, kBase+24..31}
__device__ __forceinline__ v16bf load_b_frag(const unsigned short* __restrict__ baseT,
                                             int ld, int nBase, int kBase, int lane) {
  int n  = lane & 15;
  int kh = (lane >> 4) * 8;
  const unsigned short* p = baseT + (size_t)(nBase + n) * ld + kBase + kh;
  Frag16 f;
  f.q[0] = *(const uint4*)(p);       // K offsets kh .. kh+7
  f.q[1] = *(const uint4*)(p + 16);  // K offsets 16+kh .. 16+kh+7
  return f.v;
}

// ---------------- 1) router logits: one wave per token ----------------
__global__ __launch_bounds__(256) void router_kernel(const float* __restrict__ x,
                                                     const float* __restrict__ wr,
                                                     float* __restrict__ logits) {
  int tok  = (int)((blockIdx.x * blockDim.x + threadIdx.x) >> 5);  // wave id = token
  int lane = threadIdx.x & 31;
  if (tok >= BATCH * SEQ) return;
  const float4* xr = (const float4*)(x + (size_t)tok * DMODEL);
  const float4* wv = (const float4*)wr;
  float acc = 0.f;
#pragma unroll
  for (int j = 0; j < 8; ++j) {               // 256 float4 per row / 32 lanes
    int e = j * 32 + lane;
    float4 a = xr[e];
    float4 b = wv[e];
    acc += a.x * b.x + a.y * b.y + a.z * b.z + a.w * b.w;
  }
#pragma unroll
  for (int off = 16; off > 0; off >>= 1) acc += __shfl_down(acc, off, 32);
  if (lane == 0) logits[tok] = acc;
}

// ---------------- 2) per-batch top-k select + softmax weights ----------------
// one workgroup (1024 threads) per batch; each thread owns 4 consecutive tokens
__global__ __launch_bounds__(1024) void select_topk_kernel(const float* __restrict__ logits,
                                                           int* __restrict__ sel,
                                                           float* __restrict__ rw,
                                                           int* __restrict__ flags) {
  __shared__ unsigned sScan[1024];
  __shared__ float    sRed[1024];
  __shared__ unsigned sCnt;

  const int b   = blockIdx.x;
  const int tid = threadIdx.x;

  float    vals[4];
  unsigned keys[4];
#pragma unroll
  for (int i = 0; i < 4; ++i) {
    int t = tid * 4 + i;
    float v = logits[b * SEQ + t];
    vals[i] = v;
    unsigned u = __float_as_uint(v);
    keys[i] = (u & 0x80000000u) ? ~u : (u | 0x80000000u);  // monotonic key
  }

  // radix-select the KTOP-th largest key
  unsigned prefix = 0, remaining = KTOP;
  for (int bit = 31; bit >= 0; --bit) {
    unsigned cand  = prefix | (1u << bit);
    unsigned hmask = ~((1u << bit) - 1u);
    if (tid == 0) sCnt = 0;
    __syncthreads();
    unsigned c = 0;
#pragma unroll
    for (int i = 0; i < 4; ++i) c += ((keys[i] & hmask) == cand) ? 1u : 0u;
    if (c) atomicAdd(&sCnt, c);
    __syncthreads();
    unsigned tot = sCnt;
    __syncthreads();
    if (tot >= remaining) prefix = cand;
    else remaining -= tot;
  }
  const unsigned T = prefix;  // key of the k-th largest logit

  // count strictly-greater to know how many ties to take (lowest index first)
  if (tid == 0) sCnt = 0;
  __syncthreads();
  {
    unsigned c = 0;
#pragma unroll
    for (int i = 0; i < 4; ++i) c += (keys[i] > T) ? 1u : 0u;
    if (c) atomicAdd(&sCnt, c);
  }
  __syncthreads();
  const unsigned tieNeed = KTOP - sCnt;
  __syncthreads();

  // scan #1: rank among ties (token-ascending)
  unsigned eqf[4];
  unsigned eqSum = 0;
#pragma unroll
  for (int i = 0; i < 4; ++i) { eqf[i] = (keys[i] == T) ? 1u : 0u; eqSum += eqf[i]; }
  sScan[tid] = eqSum;
  __syncthreads();
  for (int off = 1; off < 1024; off <<= 1) {
    unsigned v2 = (tid >= off) ? sScan[tid - off] : 0u;
    __syncthreads();
    sScan[tid] += v2;
    __syncthreads();
  }
  unsigned eqExcl = sScan[tid] - eqSum;
  __syncthreads();

  unsigned selected[4];
  {
    unsigned run = eqExcl;
#pragma unroll
    for (int i = 0; i < 4; ++i) {
      unsigned isSel = (keys[i] > T) ? 1u : ((eqf[i] && run < tieNeed) ? 1u : 0u);
      run += eqf[i];
      selected[i] = isSel;
    }
  }

  // softmax over selected logits: global max is always selected (top-1)
  float lm = -3.4e38f;
#pragma unroll
  for (int i = 0; i < 4; ++i) lm = fmaxf(lm, vals[i]);
  sRed[tid] = lm;
  __syncthreads();
  for (int off = 512; off > 0; off >>= 1) {
    if (tid < off) sRed[tid] = fmaxf(sRed[tid], sRed[tid + off]);
    __syncthreads();
  }
  const float mx = sRed[0];
  __syncthreads();

  float ls = 0.f;
#pragma unroll
  for (int i = 0; i < 4; ++i)
    if (selected[i]) ls += __expf(vals[i] - mx);
  sRed[tid] = ls;
  __syncthreads();
  for (int off = 512; off > 0; off >>= 1) {
    if (tid < off) sRed[tid] += sRed[tid + off];
    __syncthreads();
  }
  const float denom = sRed[0];
  __syncthreads();

  // scan #2: compacted position (ascending token order) and writes
  unsigned selSum = selected[0] + selected[1] + selected[2] + selected[3];
  sScan[tid] = selSum;
  __syncthreads();
  for (int off = 1; off < 1024; off <<= 1) {
    unsigned v2 = (tid >= off) ? sScan[tid - off] : 0u;
    __syncthreads();
    sScan[tid] += v2;
    __syncthreads();
  }
  unsigned pos = sScan[tid] - selSum;
#pragma unroll
  for (int i = 0; i < 4; ++i) {
    int t = tid * 4 + i;
    flags[b * SEQ + t] = (int)selected[i];
    if (selected[i]) {
      sel[b * KTOP + pos] = t;
      rw[b * KTOP + pos]  = __expf(vals[i] - mx) / denom;
      ++pos;
    }
  }
}

// ---------------- 3) weight transpose+convert: src[K][N] f32 -> dst[N][K] bf16 ----------------
template <int K, int N>
__global__ __launch_bounds__(256) void convert_T_kernel(const float* __restrict__ src,
                                                        unsigned short* __restrict__ dst) {
  size_t total  = (size_t)K * N;
  size_t stride = (size_t)gridDim.x * blockDim.x;
  for (size_t i = blockIdx.x * (size_t)blockDim.x + threadIdx.x; i < total; i += stride) {
    size_t n = i / K, c = i % K;
    dst[i] = f2bf(src[c * (size_t)N + n]);
  }
}

// ---------------- 4) gather selected rows -> bf16 fx [MTOT x DMODEL] ----------------
__global__ __launch_bounds__(256) void gather_fx_kernel(const float* __restrict__ x,
                                                        const int* __restrict__ sel,
                                                        unsigned short* __restrict__ fxb) {
  int m = blockIdx.x;               // selected-row index
  int b = m >> 11;                  // / KTOP
  int token = sel[m];
  const float4* src = (const float4*)(x + ((size_t)b * SEQ + token) * DMODEL);
  float4 v = src[threadIdx.x];      // 256 threads * 4 = 1024
  unsigned lo = (unsigned)f2bf(v.x) | ((unsigned)f2bf(v.y) << 16);
  unsigned hi = (unsigned)f2bf(v.z) | ((unsigned)f2bf(v.w) << 16);
  ((uint2*)(fxb + (size_t)m * DMODEL))[threadIdx.x] = make_uint2(lo, hi);
}

// ---------------- 5) GEMM1: H = silu(fx @ W1), bf16 WMMA, f32 accum ----------------
// grid (MTOT/128, DFF/256), 8 waves per WG arranged 2M x 4N, wave tile 64x64
__global__ __launch_bounds__(256) void gemm1_kernel(const unsigned short* __restrict__ fxb,
                                                    const unsigned short* __restrict__ w1t,
                                                    unsigned short* __restrict__ H) {
  const int lane  = threadIdx.x & 31;
  const int w     = threadIdx.x >> 5;
  const int mBase = blockIdx.x * 128 + (w & 1) * 64;
  const int nBase = blockIdx.y * 256 + (w >> 1) * 64;

  v8f acc[4][4] = {};
  for (int k = 0; k < DMODEL; k += 32) {
    v16bf a0 = load_a_frag(fxb, DMODEL, mBase,      k, lane);
    v16bf a1 = load_a_frag(fxb, DMODEL, mBase + 16, k, lane);
    v16bf a2 = load_a_frag(fxb, DMODEL, mBase + 32, k, lane);
    v16bf a3 = load_a_frag(fxb, DMODEL, mBase + 48, k, lane);
#pragma unroll
    for (int j = 0; j < 4; ++j) {
      v16bf bj = load_b_frag(w1t, DMODEL, nBase + j * 16, k, lane);
      acc[0][j] = __builtin_amdgcn_wmma_f32_16x16x32_bf16(false, a0, false, bj,
                                                          (short)0, acc[0][j], false, false);
      acc[1][j] = __builtin_amdgcn_wmma_f32_16x16x32_bf16(false, a1, false, bj,
                                                          (short)0, acc[1][j], false, false);
      acc[2][j] = __builtin_amdgcn_wmma_f32_16x16x32_bf16(false, a2, false, bj,
                                                          (short)0, acc[2][j], false, false);
      acc[3][j] = __builtin_amdgcn_wmma_f32_16x16x32_bf16(false, a3, false, bj,
                                                          (short)0, acc[3][j], false, false);
    }
  }

  const int n  = lane & 15;
  const int hi = lane >> 4;
#pragma unroll
  for (int i = 0; i < 4; ++i)
#pragma unroll
    for (int j = 0; j < 4; ++j)
#pragma unroll
      for (int g = 0; g < 8; ++g) {
        int row = mBase + i * 16 + g + 8 * hi;
        int col = nBase + j * 16 + n;
        float v = acc[i][j][g];
        float s = v / (1.f + __expf(-v));      // silu
        H[(size_t)row * DFF + col] = f2bf(s);
      }
}

// ---------------- 6) GEMM2: out = x + (H @ W2) * r_w, scattered to d_out ----------------
// grid (MTOT/128, DMODEL/256), wave tile 64x64
__global__ __launch_bounds__(256) void gemm2_kernel(const unsigned short* __restrict__ H,
                                                    const unsigned short* __restrict__ w2t,
                                                    const float* __restrict__ x,
                                                    const int* __restrict__ sel,
                                                    const float* __restrict__ rw,
                                                    float* __restrict__ out) {
  const int lane  = threadIdx.x & 31;
  const int w     = threadIdx.x >> 5;
  const int mBase = blockIdx.x * 128 + (w & 1) * 64;
  const int nBase = blockIdx.y * 256 + (w >> 1) * 64;

  v8f acc[4][4] = {};
  for (int k = 0; k < DFF; k += 32) {
    v16bf a0 = load_a_frag(H, DFF, mBase,      k, lane);
    v16bf a1 = load_a_frag(H, DFF, mBase + 16, k, lane);
    v16bf a2 = load_a_frag(H, DFF, mBase + 32, k, lane);
    v16bf a3 = load_a_frag(H, DFF, mBase + 48, k, lane);
#pragma unroll
    for (int j = 0; j < 4; ++j) {
      v16bf bj = load_b_frag(w2t, DFF, nBase + j * 16, k, lane);
      acc[0][j] = __builtin_amdgcn_wmma_f32_16x16x32_bf16(false, a0, false, bj,
                                                          (short)0, acc[0][j], false, false);
      acc[1][j] = __builtin_amdgcn_wmma_f32_16x16x32_bf16(false, a1, false, bj,
                                                          (short)0, acc[1][j], false, false);
      acc[2][j] = __builtin_amdgcn_wmma_f32_16x16x32_bf16(false, a2, false, bj,
                                                          (short)0, acc[2][j], false, false);
      acc[3][j] = __builtin_amdgcn_wmma_f32_16x16x32_bf16(false, a3, false, bj,
                                                          (short)0, acc[3][j], false, false);
    }
  }

  const int n  = lane & 15;
  const int hi = lane >> 4;
#pragma unroll
  for (int i = 0; i < 4; ++i) {
#pragma unroll
    for (int g = 0; g < 8; ++g) {
      int row = mBase + i * 16 + g + 8 * hi;
      int b = row >> 11;            // / KTOP
      int token = sel[row];
      float scale = rw[row];
      size_t rowOff = ((size_t)b * SEQ + token) * DMODEL;
#pragma unroll
      for (int j = 0; j < 4; ++j) {
        int col = nBase + j * 16 + n;
        size_t idx = rowOff + col;
        out[idx] = x[idx] + acc[i][j][g] * scale;
      }
    }
  }
}

// ---------------- 7) aux loss ----------------
__global__ __launch_bounds__(1024) void aux_loss_kernel(const float* __restrict__ logits,
                                                        const int* __restrict__ flags,
                                                        float* __restrict__ outLoss) {
  __shared__ float red[1024];
  float s = 0.f;
  for (int i = threadIdx.x; i < BATCH * SEQ; i += 1024) {
    float l = logits[i];
    float t = (float)flags[i];
    s += fmaxf(l, 0.f) - l * t + log1pf(__expf(-fabsf(l)));
  }
  red[threadIdx.x] = s;
  __syncthreads();
  for (int off = 512; off > 0; off >>= 1) {
    if (threadIdx.x < off) red[threadIdx.x] += red[threadIdx.x + off];
    __syncthreads();
  }
  if (threadIdx.x == 0) outLoss[0] = red[0] / (float)(BATCH * SEQ);
}

// ---------------- launch ----------------
extern "C" void kernel_launch(void* const* d_in, const int* in_sizes, int n_in,
                              void* d_out, int out_size, void* d_ws, size_t ws_size,
                              hipStream_t stream) {
  const float* x  = (const float*)d_in[0];
  // d_in[1] = mask (unused by the reference math)
  const float* Wr = (const float*)d_in[2];
  const float* W1 = (const float*)d_in[3];
  const float* W2 = (const float*)d_in[4];
  float* out = (float*)d_out;

  char* ws = (char*)d_ws;
  float* logits = (float*)(ws);                       // 64 KB
  int*   sel    = (int*)  (ws + (64 << 10));          // 32 KB
  float* rw     = (float*)(ws + (96 << 10));          // 32 KB
  int*   flags  = (int*)  (ws + (128 << 10));         // 64 KB
  unsigned short* fxb  = (unsigned short*)(ws + (192 << 10));     // 16 MB
  unsigned short* w1t  = fxb + (size_t)MTOT * DMODEL;             //  8 MB
  unsigned short* w2t  = w1t + (size_t)DFF * DMODEL;              //  8 MB
  unsigned short* Hbuf = w2t + (size_t)DMODEL * DFF;              // 64 MB

  // output starts as a copy of x; gemm2 scatters the processed rows over it
  hipMemcpyAsync(d_out, d_in[0], (size_t)BATCH * SEQ * DMODEL * sizeof(float),
                 hipMemcpyDeviceToDevice, stream);

  router_kernel<<<(BATCH * SEQ) / 8, 256, 0, stream>>>(x, Wr, logits);
  select_topk_kernel<<<BATCH, 1024, 0, stream>>>(logits, sel, rw, flags);
  convert_T_kernel<DMODEL, DFF><<<4096, 256, 0, stream>>>(W1, w1t);   // W1[K=1024][N=4096] -> [N][K]
  convert_T_kernel<DFF, DMODEL><<<4096, 256, 0, stream>>>(W2, w2t);   // W2[K=4096][N=1024] -> [N][K]
  gather_fx_kernel<<<MTOT, 256, 0, stream>>>(x, sel, fxb);
  gemm1_kernel<<<dim3(MTOT / 128, DFF / 256), 256, 0, stream>>>(fxb, w1t, Hbuf);
  gemm2_kernel<<<dim3(MTOT / 128, DMODEL / 256), 256, 0, stream>>>(Hbuf, w2t, x, sel, rw, out);
  aux_loss_kernel<<<1, 1024, 0, stream>>>(logits, flags, out + (size_t)BATCH * SEQ * DMODEL);
}
